// LocalWindowAttention_36807869727158
// MI455X (gfx1250) — matrix operand order repaired and checked by reading, not compiled, not run on hardware
//
#include <hip/hip_runtime.h>
#include <hip/hip_bf16.h>

// ---------------------------------------------------------------------------
// LocalWindowAttention forward for MI455X (gfx1250), wave32 + WMMA f16.
// Pipeline:
//   K0a: weights -> f16 (ws)       K0b: mask nearest-downsample -> rm (d_out)
//   K0c: LN(query_tokens) -> q16   K2 : per-window 2-layer transformer (LDS-resident)
//   K3 : cross-attn logits (WMMA)  K4 : row softmax + entropy rows
//   K5a: ev split-K partials (WMMA) K5b: deterministic reduce + LN -> ev
//   K6 : entropy mean -> ent
// ---------------------------------------------------------------------------

typedef __attribute__((ext_vector_type(16))) _Float16 v16h;
typedef __attribute__((ext_vector_type(8)))  float    v8f;
typedef __attribute__((ext_vector_type(4)))  int      v4i;

#define DEV __device__ __forceinline__

DEV v8f zero8() { v8f z = {0.f,0.f,0.f,0.f,0.f,0.f,0.f,0.f}; return z; }

DEV v8f wmma16(v16h a, v16h b, v8f c) {
  // D = A(16x32 f16) * B(32x16 f16) + C(f32)
  return __builtin_amdgcn_wmma_f32_16x16x32_f16(false, a, false, b, (short)0, c,
                                                false, false);
}

// A fragment: row-major [M][K], tile (row0, k0), per ISA 16-bit A layout:
// lanes 0-15 hold K {0..7,16..23}, lanes 16-31 hold K {8..15,24..31}.
DEV v16h loadA(const _Float16* base, int ld, int row0, int k0) {
  int lane = threadIdx.x & 31;
  const _Float16* p = base + (row0 + (lane & 15)) * ld + k0 + ((lane >> 4) << 3);
  v16h f;
  reinterpret_cast<v4i*>(&f)[0] = *reinterpret_cast<const v4i*>(p);
  reinterpret_cast<v4i*>(&f)[1] = *reinterpret_cast<const v4i*>(p + 16);
  return f;
}

// B fragment from B^T stored row-major as [N][K] (i.e. weight W[out][in]):
// lanes 0-15: col N=lane, K {0..7,16..23}; lanes 16-31: K {8..15,24..31}.
DEV v16h loadB(const _Float16* baseT, int ld, int col0, int k0) {
  int lane = threadIdx.x & 31;
  const _Float16* p = baseT + (col0 + (lane & 15)) * ld + k0 + ((lane >> 4) << 3);
  v16h f;
  reinterpret_cast<v4i*>(&f)[0] = *reinterpret_cast<const v4i*>(p);
  reinterpret_cast<v4i*>(&f)[1] = *reinterpret_cast<const v4i*>(p + 16);
  return f;
}

DEV float geluf(float x) { return 0.5f * x * (1.f + erff(x * 0.7071067811865475f)); }

// mean / rstd over 256-dim rows of a [64][256] f32 LDS tile; 4 lanes per row.
DEV void token_stats(const float* tok, float* s_mean, float* s_rstd) {
  int t = threadIdx.x >> 2, part = threadIdx.x & 3;
  const float* row = tok + t * 256 + part * 64;
  float s = 0.f, ss = 0.f;
  for (int j = 0; j < 64; ++j) { float v = row[j]; s += v; ss += v * v; }
  s  += __shfl_xor(s, 1);  ss += __shfl_xor(ss, 1);
  s  += __shfl_xor(s, 2);  ss += __shfl_xor(ss, 2);
  if (part == 0) {
    float m = s * (1.f / 256.f);
    float var = ss * (1.f / 256.f) - m * m;
    s_mean[t] = m;
    s_rstd[t] = rsqrtf(var + 1e-5f);
  }
}

// f16 weight arena offsets (in halfs)
#define OFF_INW   0u         // [2][768][256]
#define OFF_OUTW  393216u    // [2][256][256]
#define OFF_W1    524288u    // [2][1024][256]
#define OFF_W2    1048576u   // [2][256][1024]
#define OFF_MEW2  1572864u   // [2][256][256]
#define W16_TOTAL 1703936u

// ----------------------------- prep kernels --------------------------------
__global__ __launch_bounds__(256) void w16_kernel(
    const float* __restrict__ inw, const float* __restrict__ outw,
    const float* __restrict__ w1,  const float* __restrict__ w2,
    const float* __restrict__ mew2, _Float16* __restrict__ w16) {
  for (unsigned i = blockIdx.x * 256u + threadIdx.x; i < W16_TOTAL;
       i += gridDim.x * 256u) {
    float v;
    if (i < OFF_OUTW)      v = inw[i];
    else if (i < OFF_W1)   v = outw[i - OFF_OUTW];
    else if (i < OFF_W2)   v = w1[i - OFF_W1];
    else if (i < OFF_MEW2) v = w2[i - OFF_W2];
    else                   v = mew2[i - OFF_MEW2];
    w16[i] = (_Float16)v;
  }
}

__global__ __launch_bounds__(256) void rm_kernel(const float* __restrict__ mask,
                                                 float* __restrict__ rm) {
  int i = blockIdx.x * 256 + threadIdx.x;   // 4*128*128
  if (i < 65536) {
    int b = i >> 14, rem = i & 16383, y = rem >> 7, x = rem & 127;
    // nearest: src = floor(dst * 512/128) = 4*dst
    rm[i] = mask[((size_t)b * 512 + y * 4) * 512 + x * 4];
  }
}

__global__ __launch_bounds__(256) void qln_kernel(
    const float* __restrict__ qt, const float* __restrict__ qg,
    const float* __restrict__ qb, _Float16* __restrict__ q16) {
  __shared__ float red[256], red2[256];
  int row = blockIdx.x, j = threadIdx.x;
  float v = qt[row * 256 + j];
  red[j] = v; red2[j] = v * v; __syncthreads();
  for (int s = 128; s > 0; s >>= 1) {
    if (j < s) { red[j] += red[j + s]; red2[j] += red2[j + s]; }
    __syncthreads();
  }
  float m = red[0] / 256.f, var = red2[0] / 256.f - m * m;
  float rstd = rsqrtf(var + 1e-5f);
  q16[row * 256 + j] = (_Float16)((v - m) * rstd * qg[j] + qb[j]);
}

// --------------------------- window transformer ----------------------------
// Dynamic LDS layout (bytes):
//  [0,65536)        tok f32 [64][256]
//  [65536,65792)    mean[64]   [65792,66048) rstd[64]   [66048,66304) mtok[64]
//  M = 66560:
//    A  = M+0       f16 [64*256]  : x / o
//    Bq = M+32768   f16           : H[64*256] / qkv[64*768] / hidden[64*512]
//    C  = M+131072  : vT f16 [256*64] | logits f32 [64*64] | a16 f16 [64*64]
#define SMEM_WIN (66560 + 131072 + 57344)   // 254976 B  (<= 320KB/WGP)

__global__ __launch_bounds__(256) void win_kernel(
    const float* __restrict__ fused, const float* __restrict__ rm,
    const _Float16* __restrict__ w16,
    const float* __restrict__ n1g,  const float* __restrict__ n1b,
    const float* __restrict__ inb,  const float* __restrict__ outb,
    const float* __restrict__ ga,   const float* __restrict__ n2g,
    const float* __restrict__ n2b,  const float* __restrict__ b1,
    const float* __restrict__ b2,   const float* __restrict__ gm,
    const float* __restrict__ mew1, const float* __restrict__ meb1,
    const float* __restrict__ meb2,
    const float* __restrict__ tin_g,  const float* __restrict__ tin_b,
    const float* __restrict__ tout_g, const float* __restrict__ tout_b,
    const float* __restrict__ feat_g, const float* __restrict__ feat_b,
    float* __restrict__ refined, _Float16* __restrict__ flat16,
    _Float16* __restrict__ flatT16) {
  extern __shared__ char smem[];
  float* tok    = (float*)smem;
  float* s_mean = (float*)(smem + 65536);
  float* s_rstd = (float*)(smem + 65792);
  float* s_mtok = (float*)(smem + 66048);
  char* M = smem + 66560;
  _Float16* xA  = (_Float16*)M;
  _Float16* xB  = (_Float16*)(M + 32768);
  _Float16* vT  = (_Float16*)(M + 131072);          // [256][64] (head*32+dh, token)
  float*    slog = (float*)(M + 131072 + 32768);    // [64][64]
  _Float16* a16  = (_Float16*)(M + 131072 + 49152); // [64][64]

  const int w = blockIdx.x;
  const int b = w >> 8, wr = (w >> 4) & 15, wc = w & 15;
  const int tid = threadIdx.x, wid = tid >> 5, lane = tid & 31;

  // ---- gather window [64 tok][256 ch] from [B,C,H,W]
  for (int it = 0; it < 64; ++it) {
    int c = (tid >> 6) + 4 * it;
    int r = (tid >> 3) & 7, cc = tid & 7;
    tok[(r * 8 + cc) * 256 + c] =
        fused[(((size_t)b * 256 + c) * 128 + wr * 8 + r) * 128 + wc * 8 + cc];
  }
  if (tid < 64) {  // mask token = rm at the token's spatial position
    int r = tid >> 3, cc = tid & 7;
    s_mtok[tid] = rm[((size_t)b * 128 + wr * 8 + r) * 128 + wc * 8 + cc];
  }
  __syncthreads();
  token_stats(tok, s_mean, s_rstd);
  __syncthreads();
  for (int idx = tid; idx < 16384; idx += 256) {   // tokens = LN_tin(window)
    int t = idx >> 8, j = idx & 255;
    tok[idx] = (tok[idx] - s_mean[t]) * s_rstd[t] * tin_g[j] + tin_b[j];
  }
  __syncthreads();

  for (int l = 0; l < 2; ++l) {
    const _Float16* inwl  = w16 + OFF_INW  + (unsigned)l * 196608u;
    const _Float16* outwl = w16 + OFF_OUTW + (unsigned)l * 65536u;
    const _Float16* w1l   = w16 + OFF_W1   + (unsigned)l * 262144u;
    const _Float16* w2l   = w16 + OFF_W2   + (unsigned)l * 262144u;
    const _Float16* mew2l = w16 + OFF_MEW2 + (unsigned)l * 65536u;

    // LN1 stats + mask-embed hidden H = gelu(mtok*mew1+meb1)  -> xB
    token_stats(tok, s_mean, s_rstd);
    for (int idx = tid; idx < 16384; idx += 256) {
      int t = idx >> 8, k = idx & 255;
      xB[idx] = (_Float16)geluf(s_mtok[t] * mew1[l * 256 + k] + meb1[l * 256 + k]);
    }
    __syncthreads();

    // x = LN1(tok) + H @ mew2^T + meb2  -> xA (f16)
    for (int tt = wid; tt < 64; tt += 8) {
      int mt = tt >> 4, nt = tt & 15;
      v8f acc = zero8();
      for (int ks = 0; ks < 8; ++ks)
        acc = wmma16(loadA(xB, 256, mt * 16, ks * 32),
                     loadB(mew2l, 256, nt * 16, ks * 32), acc);
      int j = nt * 16 + (lane & 15);
      int m0 = mt * 16 + ((lane >> 4) << 3);
      float g = n1g[l * 256 + j], bb = n1b[l * 256 + j], mb = meb2[l * 256 + j];
      for (int r = 0; r < 8; ++r) {
        int m = m0 + r;
        float ln = (tok[m * 256 + j] - s_mean[m]) * s_rstd[m] * g + bb;
        xA[m * 256 + j] = (_Float16)(ln + acc[r] + mb);
      }
    }
    __syncthreads();

    // qkv = x @ inw^T + inb  -> xB[64][768];  v also transposed -> vT
    for (int tt = wid; tt < 192; tt += 8) {
      int mt = tt / 48, nt = tt % 48;
      v8f acc = zero8();
      for (int ks = 0; ks < 8; ++ks)
        acc = wmma16(loadA(xA, 256, mt * 16, ks * 32),
                     loadB(inwl, 256, nt * 16, ks * 32), acc);
      int j = nt * 16 + (lane & 15);
      int m0 = mt * 16 + ((lane >> 4) << 3);
      float bias = inb[l * 768 + j];
      for (int r = 0; r < 8; ++r) {
        int m = m0 + r;
        float v = acc[r] + bias;
        xB[m * 768 + j] = (_Float16)v;
        if (j >= 512) vT[(j - 512) * 64 + m] = (_Float16)v;
      }
    }
    __syncthreads();

    // attention (8 heads, dh=32), o -> xA
    for (int h = 0; h < 8; ++h) {
      for (int tt = wid; tt < 16; tt += 8) {   // logits = q k^T / sqrt(32)
        int mt = tt >> 2, nt = tt & 3;
        v8f acc = zero8();
        acc = wmma16(loadA(xB, 768, mt * 16, h * 32),
                     loadB(xB, 768, nt * 16, 256 + h * 32), acc);
        int n = nt * 16 + (lane & 15);
        int m0 = mt * 16 + ((lane >> 4) << 3);
        for (int r = 0; r < 8; ++r)
          slog[(m0 + r) * 64 + n] = acc[r] * 0.17677669529663687f;
      }
      __syncthreads();
      if (tid < 64) {                          // row softmax -> a16
        float* row = slog + tid * 64;
        float mx = row[0];
        for (int k = 1; k < 64; ++k) mx = fmaxf(mx, row[k]);
        float s = 0.f;
        for (int k = 0; k < 64; ++k) { float e = expf(row[k] - mx); row[k] = e; s += e; }
        float inv = 1.f / s;
        for (int k = 0; k < 64; ++k) a16[tid * 64 + k] = (_Float16)(row[k] * inv);
      }
      __syncthreads();
      {                                        // o = a @ v  (via vT rows)
        int mt = wid >> 1, ntl = wid & 1;
        v8f acc = zero8();
        for (int ks = 0; ks < 2; ++ks)
          acc = wmma16(loadA(a16, 64, mt * 16, ks * 32),
                       loadB(vT + h * 32 * 64, 64, ntl * 16, ks * 32), acc);
        int j = h * 32 + ntl * 16 + (lane & 15);
        int m0 = mt * 16 + ((lane >> 4) << 3);
        for (int r = 0; r < 8; ++r) xA[(m0 + r) * 256 + j] = (_Float16)acc[r];
      }
      __syncthreads();
    }

    // tok += ga * (o @ outw^T + outb)
    for (int tt = wid; tt < 64; tt += 8) {
      int mt = tt >> 4, nt = tt & 15;
      v8f acc = zero8();
      for (int ks = 0; ks < 8; ++ks)
        acc = wmma16(loadA(xA, 256, mt * 16, ks * 32),
                     loadB(outwl, 256, nt * 16, ks * 32), acc);
      int j = nt * 16 + (lane & 15);
      int m0 = mt * 16 + ((lane >> 4) << 3);
      float g = ga[l * 256 + j], ob = outb[l * 256 + j];
      for (int r = 0; r < 8; ++r) tok[(m0 + r) * 256 + j] += g * (acc[r] + ob);
    }
    __syncthreads();

    // MLP: tok += gm * (gelu(LN2(tok) @ w1^T + b1) @ w2^T + b2)
    token_stats(tok, s_mean, s_rstd);
    __syncthreads();
    for (int idx = tid; idx < 16384; idx += 256) {
      int t = idx >> 8, j = idx & 255;
      xA[idx] = (_Float16)((tok[idx] - s_mean[t]) * s_rstd[t] * n2g[l * 256 + j] +
                           n2b[l * 256 + j]);
    }
    __syncthreads();
    v8f oacc[8];
    for (int i = 0; i < 8; ++i) oacc[i] = zero8();
    for (int c = 0; c < 2; ++c) {              // hidden 1024 in 2 chunks of 512
      for (int tt = wid; tt < 128; tt += 8) {
        int mt = tt >> 5, nt = tt & 31;
        v8f acc = zero8();
        for (int ks = 0; ks < 8; ++ks)
          acc = wmma16(loadA(xA, 256, mt * 16, ks * 32),
                       loadB(w1l + c * 512 * 256, 256, nt * 16, ks * 32), acc);
        int j = nt * 16 + (lane & 15);
        int m0 = mt * 16 + ((lane >> 4) << 3);
        float bb = b1[l * 1024 + c * 512 + j];
        for (int r = 0; r < 8; ++r)
          xB[(m0 + r) * 512 + j] = (_Float16)geluf(acc[r] + bb);
      }
      __syncthreads();
      for (int i = 0; i < 8; ++i) {
        int tt = wid + i * 8;
        int mt = tt >> 4, nt = tt & 15;
        for (int ks = 0; ks < 16; ++ks)
          oacc[i] = wmma16(loadA(xB, 512, mt * 16, ks * 32),
                           loadB(w2l, 1024, nt * 16, c * 512 + ks * 32), oacc[i]);
      }
      __syncthreads();
    }
    for (int i = 0; i < 8; ++i) {
      int tt = wid + i * 8;
      int mt = tt >> 4, nt = tt & 15;
      int j = nt * 16 + (lane & 15);
      int m0 = mt * 16 + ((lane >> 4) << 3);
      float g = gm[l * 256 + j], bb = b2[l * 256 + j];
      for (int r = 0; r < 8; ++r) tok[(m0 + r) * 256 + j] += g * (oacc[i][r] + bb);
    }
    __syncthreads();
  }

  // final LN_tout in place -> refined values
  token_stats(tok, s_mean, s_rstd);
  __syncthreads();
  for (int idx = tid; idx < 16384; idx += 256) {
    int t = idx >> 8, j = idx & 255;
    tok[idx] = (tok[idx] - s_mean[t]) * s_rstd[t] * tout_g[j] + tout_b[j];
  }
  __syncthreads();
  token_stats(tok, s_mean, s_rstd);  // stats of refined rows for feat-LN
  __syncthreads();
  for (int idx = tid; idx < 16384; idx += 256) {
    int t = idx >> 8, j = idx & 255;
    float v = tok[idx];
    int r = t >> 3, cc = t & 7;
    int hh = wr * 8 + r, ww2 = wc * 8 + cc;
    int n = hh * 128 + ww2;
    refined[(((size_t)b * 256 + j) * 128 + hh) * 128 + ww2] = v;
    float f = (v - s_mean[t]) * s_rstd[t] * feat_g[j] + feat_b[j];
    flat16[((size_t)b * 16384 + n) * 256 + j] = (_Float16)f;
    flatT16[((size_t)b * 256 + j) * 16384 + n] = (_Float16)f;
  }
}

// --------------------------- cross attention -------------------------------
__global__ __launch_bounds__(256) void xlogits_kernel(
    const _Float16* __restrict__ q16, const _Float16* __restrict__ flat16,
    const float* __restrict__ rm, float* __restrict__ attn_out) {
  int b = blockIdx.y;
  int wid = threadIdx.x >> 5;
  int mt = wid & 3, ntl = wid >> 2;
  int n0 = (blockIdx.x * 2 + ntl) * 16;
  const _Float16* fb = flat16 + (size_t)b * 16384 * 256;
  v8f acc = zero8();
  for (int ks = 0; ks < 8; ++ks)
    acc = wmma16(loadA(q16, 256, mt * 16, ks * 32), loadB(fb, 256, n0, ks * 32), acc);
  int lane = threadIdx.x & 31;
  int n = n0 + (lane & 15);
  int m0 = mt * 16 + ((lane >> 4) << 3);
  float bias = rm[(size_t)b * 16384 + n] * 10.f - 5.f;  // fm*5 + (1-fm)*(-5)
  for (int r = 0; r < 8; ++r)
    attn_out[((size_t)(b * 64 + m0 + r)) * 16384 + n] = acc[r] * 0.0625f + bias;
}

__global__ __launch_bounds__(256) void softmax_kernel(
    float* __restrict__ attn, _Float16* __restrict__ attn16,
    float* __restrict__ entrow) {
  __shared__ float red[256];
  int row = blockIdx.x;  // b*64+t
  float* p = attn + (size_t)row * 16384;
  int tid = threadIdx.x;
  float mx = -3.0e38f;
  for (int i = tid; i < 16384; i += 256) mx = fmaxf(mx, p[i]);
  red[tid] = mx; __syncthreads();
  for (int s = 128; s > 0; s >>= 1) {
    if (tid < s) red[tid] = fmaxf(red[tid], red[tid + s]);
    __syncthreads();
  }
  mx = red[0]; __syncthreads();
  float sum = 0.f;
  for (int i = tid; i < 16384; i += 256) sum += expf(p[i] - mx);
  red[tid] = sum; __syncthreads();
  for (int s = 128; s > 0; s >>= 1) {
    if (tid < s) red[tid] += red[tid + s];
    __syncthreads();
  }
  float inv = 1.f / red[0]; __syncthreads();
  float ent = 0.f;
  for (int i = tid; i < 16384; i += 256) {
    float v = expf(p[i] - mx) * inv;
    p[i] = v;
    attn16[(size_t)row * 16384 + i] = (_Float16)v;
    float pc = fmaxf(v, 1e-8f);
    ent -= pc * logf(pc);
  }
  red[tid] = ent; __syncthreads();
  for (int s = 128; s > 0; s >>= 1) {
    if (tid < s) red[tid] += red[tid + s];
    __syncthreads();
  }
  if (tid == 0) entrow[row] = red[0];
}

__global__ __launch_bounds__(256) void evpart_kernel(
    const _Float16* __restrict__ attn16, const _Float16* __restrict__ flatT16,
    float* __restrict__ evpart) {
  int sl = blockIdx.x;  // k-slice 0..15 (each covers 1024 of N=16384)
  int b = blockIdx.y;
  int wid = threadIdx.x >> 5;
  const _Float16* A  = attn16  + (size_t)b * 64 * 16384;
  const _Float16* Bt = flatT16 + (size_t)b * 256 * 16384;
  int kbase = sl * 1024;
  int lane = threadIdx.x & 31;
  for (int i = 0; i < 8; ++i) {
    int tt = wid + i * 8;
    int mt = tt >> 4, nt = tt & 15;
    v8f acc = zero8();
    for (int ks = 0; ks < 32; ++ks)
      acc = wmma16(loadA(A, 16384, mt * 16, kbase + ks * 32),
                   loadB(Bt, 16384, nt * 16, kbase + ks * 32), acc);
    int n = nt * 16 + (lane & 15);
    int m0 = mt * 16 + ((lane >> 4) << 3);
    for (int r = 0; r < 8; ++r)
      evpart[(((size_t)sl * 4 + b) * 64 + m0 + r) * 256 + n] = acc[r];
  }
}

__global__ __launch_bounds__(256) void evln_kernel(
    const float* __restrict__ evpart, const float* __restrict__ out_g,
    const float* __restrict__ out_b, float* __restrict__ ev) {
  __shared__ float red[256], red2[256];
  int row = blockIdx.x;  // b*64+t
  int j = threadIdx.x;
  int b = row >> 6, t = row & 63;
  float s = 0.f;
  for (int sl = 0; sl < 16; ++sl)   // fixed order -> deterministic
    s += evpart[(((size_t)sl * 4 + b) * 64 + t) * 256 + j];
  red[j] = s; red2[j] = s * s; __syncthreads();
  for (int st = 128; st > 0; st >>= 1) {
    if (j < st) { red[j] += red[j + st]; red2[j] += red2[j + st]; }
    __syncthreads();
  }
  float m = red[0] / 256.f, var = red2[0] / 256.f - m * m;
  float rstd = rsqrtf(var + 1e-5f);
  ev[(size_t)row * 256 + j] = (s - m) * rstd * out_g[j] + out_b[j];
}

__global__ __launch_bounds__(64) void ent_kernel(const float* __restrict__ entrow,
                                                 float* __restrict__ ent) {
  __shared__ float red[64];
  int b = blockIdx.x, t = threadIdx.x;
  red[t] = entrow[b * 64 + t]; __syncthreads();
  for (int s = 32; s > 0; s >>= 1) {
    if (t < s) red[t] += red[t + s];
    __syncthreads();
  }
  if (t == 0) ent[b] = red[0] / 64.f;
}

// ------------------------------- launcher ----------------------------------
extern "C" void kernel_launch(void* const* d_in, const int* in_sizes, int n_in,
                              void* d_out, int out_size, void* d_ws, size_t ws_size,
                              hipStream_t stream) {
  (void)in_sizes; (void)n_in; (void)out_size; (void)ws_size;
  const float* fused = (const float*)d_in[0];
  const float* mask  = (const float*)d_in[1];
  const float* qtok  = (const float*)d_in[2];
  const float* n1g   = (const float*)d_in[3];
  const float* n1b   = (const float*)d_in[4];
  const float* inw   = (const float*)d_in[5];
  const float* inb   = (const float*)d_in[6];
  const float* outw  = (const float*)d_in[7];
  const float* outb  = (const float*)d_in[8];
  const float* ga    = (const float*)d_in[9];
  const float* n2g   = (const float*)d_in[10];
  const float* n2b   = (const float*)d_in[11];
  const float* w1    = (const float*)d_in[12];
  const float* b1    = (const float*)d_in[13];
  const float* w2    = (const float*)d_in[14];
  const float* b2    = (const float*)d_in[15];
  const float* gm    = (const float*)d_in[16];
  const float* mew1  = (const float*)d_in[17];
  const float* meb1  = (const float*)d_in[18];
  const float* mew2  = (const float*)d_in[19];
  const float* meb2  = (const float*)d_in[20];
  const float* tin_g = (const float*)d_in[21];
  const float* tin_b = (const float*)d_in[22];
  const float* tout_g= (const float*)d_in[23];
  const float* tout_b= (const float*)d_in[24];
  const float* feat_g= (const float*)d_in[25];
  const float* feat_b= (const float*)d_in[26];
  const float* out_g = (const float*)d_in[27];
  const float* out_b = (const float*)d_in[28];
  const float* q_g   = (const float*)d_in[29];
  const float* q_b   = (const float*)d_in[30];

  // outputs, flattened in return order: refined, ev, attn, ent, rm
  float* out     = (float*)d_out;
  float* refined = out;                  // 4*256*128*128
  float* ev      = out + 16777216;       // 4*64*256
  float* attn    = out + 16842752;       // 4*64*16384
  float* ent     = out + 21037056;       // 4
  float* rmout   = out + 21037060;       // 4*1*128*128

  // workspace (~96 MB)
  char* ws = (char*)d_ws;
  _Float16* w16     = (_Float16*)(ws);              // 3,407,872 B
  _Float16* q16     = (_Float16*)(ws + 3407872);    // 32,768 B
  _Float16* flat16  = (_Float16*)(ws + 3440640);    // 33,554,432 B
  _Float16* flatT16 = (_Float16*)(ws + 36995072);   // 33,554,432 B
  _Float16* attn16  = (_Float16*)(ws + 70549504);   // 8,388,608 B
  float*    evpart  = (float*)  (ws + 78938112);    // 16,777,216 B
  float*    entrow  = (float*)  (ws + 95715328);    // 1,024 B

  hipFuncSetAttribute((const void*)win_kernel,
                      hipFuncAttributeMaxDynamicSharedMemorySize, SMEM_WIN);

  w16_kernel<<<dim3(1024), dim3(256), 0, stream>>>(inw, outw, w1, w2, mew2, w16);
  rm_kernel<<<dim3(256), dim3(256), 0, stream>>>(mask, rmout);
  qln_kernel<<<dim3(64), dim3(256), 0, stream>>>(qtok, q_g, q_b, q16);
  win_kernel<<<dim3(1024), dim3(256), SMEM_WIN, stream>>>(
      fused, rmout, w16, n1g, n1b, inb, outb, ga, n2g, n2b, b1, b2, gm,
      mew1, meb1, meb2, tin_g, tin_b, tout_g, tout_b, feat_g, feat_b,
      refined, flat16, flatT16);
  xlogits_kernel<<<dim3(512, 4), dim3(256), 0, stream>>>(q16, flat16, rmout, attn);
  softmax_kernel<<<dim3(256), dim3(256), 0, stream>>>(attn, attn16, entrow);
  evpart_kernel<<<dim3(16, 4), dim3(256), 0, stream>>>(attn16, flatT16, evpart);
  evln_kernel<<<dim3(256), dim3(256), 0, stream>>>(evpart, out_g, out_b, ev);
  ent_kernel<<<dim3(4), dim3(64), 0, stream>>>(entrow, ent);
}
